// ProtoNetLayer_53188874994041
// MI455X (gfx1250) — matrix operand order repaired and checked by reading, not compiled, not run on hardware
//
#include <hip/hip_runtime.h>
#include <hip/hip_bf16.h>

// ProtoNet scoring layer for MI455X (gfx1250), bf16 WMMA pipeline.
// Shapes fixed by reference: N=16384 rows, D=1024, Dp=256, K=2048.

typedef __attribute__((ext_vector_type(16))) __bf16 v16bf;
typedef __attribute__((ext_vector_type(8)))  float  v8f;
typedef __attribute__((ext_vector_type(4)))  unsigned int u32x4;
typedef __attribute__((ext_vector_type(4)))  float  f32x4;

#define N_ROWS 16384
#define DIM    1024
#define DP     256
#define KPROTO 2048

union FragU { u32x4 u[2]; v16bf v; };

static __device__ __forceinline__ unsigned short f2bf(float f) {
  unsigned int u = __float_as_uint(f);
  u += 0x7FFFu + ((u >> 16) & 1u);     // round-to-nearest-even
  return (unsigned short)(u >> 16);
}

// ---------------- prep kernels ----------------

// Wt[dp][d] = bf16(proj[d][dp])   (transpose so B-fragments are K-contiguous)
__global__ __launch_bounds__(256) void k_transpose_w(const float* __restrict__ proj,
                                                     unsigned short* __restrict__ Wt) {
  int idx = blockIdx.x * 256 + threadIdx.x;   // 0 .. 1024*256-1
  int d  = idx >> 8;
  int dp = idx & 255;
  Wt[dp * DIM + d] = f2bf(proj[idx]);
}

// Pb = bf16(prototypes), pp[k] = sum_dp P[k][dp]^2
__global__ __launch_bounds__(256) void k_proto_prep(const float* __restrict__ P,
                                                    unsigned short* __restrict__ Pb,
                                                    float* __restrict__ pp) {
  __shared__ float red[256];
  int k = blockIdx.x;
  int t = threadIdx.x;
  float v = P[k * DP + t];
  Pb[k * DP + t] = f2bf(v);
  red[t] = v * v;
  __syncthreads();
  #pragma unroll
  for (int s = 128; s > 0; s >>= 1) {
    if (t < s) red[t] += red[t + s];
    __syncthreads();
  }
  if (t == 0) pp[k] = red[0];
}

// mW[dp] = sum_d mean[d] * proj[d][dp]   (folds the centering into the epilogue)
__global__ __launch_bounds__(256) void k_mean_proj(const float* __restrict__ mean,
                                                   const float* __restrict__ proj,
                                                   float* __restrict__ mW) {
  int dp = threadIdx.x;
  float s = 0.f;
  for (int d = 0; d < DIM; ++d) s += mean[d] * proj[d * DP + dp];
  mW[dp] = s;
}

// ---------------- GEMM1: Z = normalize(X @ W - mW) -> bf16 ----------------
// Block tile: 32 (M) x 256 (N = full Dp). 8 waves, each wave 32x32 (2x2 WMMA tiles).
__global__ __launch_bounds__(256) void
k_gemm1_proj_norm(const float* __restrict__ X, const unsigned short* __restrict__ Wt,
                  const float* __restrict__ mW, unsigned short* __restrict__ Zn) {
  __shared__ __align__(16) char smem[4096 + 32768];
  unsigned short* sA = (unsigned short*)smem;            //  32 x 64 bf16 (4 KB)
  unsigned short* sB = (unsigned short*)(smem + 4096);   // 256 x 64 bf16 (32 KB)
  float*          sZ = (float*)(smem + 4096);            // alias: 32 x 256 f32 (32 KB)
  __shared__ float sInv[32];

  const int tid  = threadIdx.x;
  const int w    = tid >> 5;
  const int lane = tid & 31;
  const int half = lane >> 4;
  const int lr   = lane & 15;
  const int n_off = w * 32;
  const int blockM = blockIdx.x;          // 512 blocks * 32 rows

  v8f acc[2][2] = {};

  for (int kt = 0; kt < DIM; kt += 64) {
    // stage A: 32x64 fp32 -> bf16 (512 float4 loads)
    #pragma unroll
    for (int i = 0; i < 2; ++i) {
      int j   = tid + i * 256;
      int row = j >> 4;               // 16 float4 per row
      int c4  = (j & 15) << 2;        // float column
      f32x4 v = *(const f32x4*)(X + (size_t)(blockM * 32 + row) * DIM + kt + c4);
      unsigned short* dst = sA + row * 64 + c4;
      dst[0] = f2bf(v.x); dst[1] = f2bf(v.y); dst[2] = f2bf(v.z); dst[3] = f2bf(v.w);
    }
    // stage B: Wt 256x64 bf16 (2048 16B copies)
    #pragma unroll
    for (int i = 0; i < 8; ++i) {
      int j   = tid + i * 256;
      int row = j >> 3;               // 8 uint4 per row
      int c   = (j & 7) << 3;         // bf16 column
      *(u32x4*)(sB + row * 64 + c) = *(const u32x4*)(Wt + (size_t)row * DIM + kt + c);
    }
    __syncthreads();

    #pragma unroll
    for (int k0 = 0; k0 < 64; k0 += 32) {
      FragU a[2], b[2];
      #pragma unroll
      for (int tm = 0; tm < 2; ++tm) {
        int m = tm * 16 + lr;
        a[tm].u[0] = *(const u32x4*)(sA + m * 64 + k0 + 8 * half);
        a[tm].u[1] = *(const u32x4*)(sA + m * 64 + k0 + 16 + 8 * half);
      }
      #pragma unroll
      for (int tn = 0; tn < 2; ++tn) {
        int n = n_off + tn * 16 + lr;
        b[tn].u[0] = *(const u32x4*)(sB + n * 64 + k0 + 8 * half);
        b[tn].u[1] = *(const u32x4*)(sB + n * 64 + k0 + 16 + 8 * half);
      }
      #pragma unroll
      for (int tm = 0; tm < 2; ++tm)
        #pragma unroll
        for (int tn = 0; tn < 2; ++tn)
          acc[tm][tn] = __builtin_amdgcn_wmma_f32_16x16x32_bf16(
              false, a[tm].v, false, b[tn].v, (short)0, acc[tm][tn], false, false);
    }
    __syncthreads();
  }

  // epilogue: Z - mW into LDS, row norms, normalized bf16 out
  #pragma unroll
  for (int tn = 0; tn < 2; ++tn) {
    int n = n_off + tn * 16 + lr;
    float mwv = mW[n];
    #pragma unroll
    for (int tm = 0; tm < 2; ++tm) {
      #pragma unroll
      for (int r = 0; r < 8; ++r) {
        int m = tm * 16 + r + 8 * half;
        sZ[m * DP + n] = acc[tm][tn][r] - mwv;
      }
    }
  }
  __syncthreads();
  if (tid < 32) {
    float s = 0.f;
    for (int j = 0; j < DP; ++j) { float v = sZ[tid * DP + j]; s += v * v; }
    float nrm = __builtin_sqrtf(s);
    sInv[tid] = 1.0f / fmaxf(nrm, 1e-12f);
  }
  __syncthreads();
  for (int i = 0; i < 32; ++i)
    Zn[(size_t)(blockM * 32 + i) * DP + tid] = f2bf(sZ[i * DP + tid] * sInv[i]);
}

// ---------------- GEMM2: scores = -sqrt(max(1 + pp - 2 Z P^T, 0)) ----------------
// Block tile 128x128, K=256 staged in two 128-wide LDS chunks (64 KB).
// 8 waves: wave_m in {0,1} (64 rows each), wave_n in {0..3} (32 cols each).
__global__ __launch_bounds__(256) void
k_gemm2_scores(const unsigned short* __restrict__ Zn, const unsigned short* __restrict__ Pb,
               const float* __restrict__ pp, float* __restrict__ out) {
  __shared__ __align__(16) char smem[65536];
  unsigned short* sA = (unsigned short*)smem;              // 128 x 128 bf16
  unsigned short* sB = (unsigned short*)(smem + 32768);    // 128 x 128 bf16

  const int tid  = threadIdx.x;
  const int w    = tid >> 5;
  const int lane = tid & 31;
  const int half = lane >> 4;
  const int lr   = lane & 15;
  const int m_off = (w & 1) * 64;
  const int n_off = (w >> 1) * 32;
  const int blockN = blockIdx.x;   // 16
  const int blockM = blockIdx.y;   // 128

  v8f acc[4][2] = {};

  #pragma unroll
  for (int kt = 0; kt < DP; kt += 128) {
    #pragma unroll
    for (int i = 0; i < 8; ++i) {
      int j   = tid + i * 256;
      int row = j >> 4;            // 16 x 16B per 128-col row
      int c   = (j & 15) << 3;     // bf16 column
      *(u32x4*)(sA + row * 128 + c) =
          *(const u32x4*)(Zn + (size_t)(blockM * 128 + row) * DP + kt + c);
      *(u32x4*)(sB + row * 128 + c) =
          *(const u32x4*)(Pb + (size_t)(blockN * 128 + row) * DP + kt + c);
    }
    __syncthreads();

    #pragma unroll
    for (int k0 = 0; k0 < 128; k0 += 32) {
      FragU a[4], b[2];
      #pragma unroll
      for (int tm = 0; tm < 4; ++tm) {
        int m = m_off + tm * 16 + lr;
        a[tm].u[0] = *(const u32x4*)(sA + m * 128 + k0 + 8 * half);
        a[tm].u[1] = *(const u32x4*)(sA + m * 128 + k0 + 16 + 8 * half);
      }
      #pragma unroll
      for (int tn = 0; tn < 2; ++tn) {
        int n = n_off + tn * 16 + lr;
        b[tn].u[0] = *(const u32x4*)(sB + n * 128 + k0 + 8 * half);
        b[tn].u[1] = *(const u32x4*)(sB + n * 128 + k0 + 16 + 8 * half);
      }
      #pragma unroll
      for (int tm = 0; tm < 4; ++tm)
        #pragma unroll
        for (int tn = 0; tn < 2; ++tn)
          acc[tm][tn] = __builtin_amdgcn_wmma_f32_16x16x32_bf16(
              false, a[tm].v, false, b[tn].v, (short)0, acc[tm][tn], false, false);
    }
    __syncthreads();
  }

  // epilogue: d2 = 1 + ||p||^2 - 2 z.p  (||z||^2 == 1 after normalization)
  #pragma unroll
  for (int tn = 0; tn < 2; ++tn) {
    int n = blockN * 128 + n_off + tn * 16 + lr;
    float ppv = pp[n];
    #pragma unroll
    for (int tm = 0; tm < 4; ++tm) {
      int mbase = blockM * 128 + m_off + tm * 16 + 8 * half;
      #pragma unroll
      for (int r = 0; r < 8; ++r) {
        float d2 = fmaxf(1.0f + ppv - 2.0f * acc[tm][tn][r], 0.0f);
        out[(size_t)(mbase + r) * KPROTO + n] = -__builtin_sqrtf(d2);
      }
    }
  }
}

// ---------------- launcher ----------------
extern "C" void kernel_launch(void* const* d_in, const int* in_sizes, int n_in,
                              void* d_out, int out_size, void* d_ws, size_t ws_size,
                              hipStream_t stream) {
  const float* X    = (const float*)d_in[0];   // [8,2048,1024]
  const float* mean = (const float*)d_in[1];   // [1024]
  const float* proj = (const float*)d_in[2];   // [1024,256]
  const float* prot = (const float*)d_in[3];   // [2048,256]
  float* out = (float*)d_out;                  // [8,2048,2048]

  char* ws = (char*)d_ws;
  unsigned short* Wt = (unsigned short*)(ws);             // 256x1024 bf16 (512 KB)
  unsigned short* Pb = (unsigned short*)(ws + 524288);    // 2048x256 bf16 (1 MB)
  float* pp          = (float*)(ws + 1572864);            // 2048 f32 (8 KB)
  float* mW          = (float*)(ws + 1581056);            // 256 f32 (1 KB)
  unsigned short* Zn = (unsigned short*)(ws + 1582080);   // 16384x256 bf16 (8 MB)

  k_transpose_w<<<(DIM * DP) / 256, 256, 0, stream>>>(proj, Wt);
  k_proto_prep<<<KPROTO, 256, 0, stream>>>(prot, Pb, pp);
  k_mean_proj<<<1, 256, 0, stream>>>(mean, proj, mW);

  k_gemm1_proj_norm<<<N_ROWS / 32, 256, 0, stream>>>(X, Wt, mW, Zn);

  dim3 g2(KPROTO / 128, N_ROWS / 128);
  k_gemm2_scores<<<g2, 256, 0, stream>>>(Zn, Pb, pp, out);
}
